// MMRecBlock_8031588843646
// MI455X (gfx1250) — matrix operand order repaired and checked by reading, not compiled, not run on hardware
//
#include <hip/hip_runtime.h>
#include <math.h>

// ---- problem constants ----
#define Bn   2
#define Tn   2048
#define Dn   512
#define Hn   8
#define HDn  64
#define Fn   2048
#define Mn   512
#define DQn  128
#define Sn   (Mn + Mn + Tn)   // 3072
#define BT   (Bn * Tn)        // 4096
#define BS   (Bn * Sn)        // 6144

typedef __attribute__((ext_vector_type(16))) __bf16 v16bf;
typedef __attribute__((ext_vector_type(8)))  float  v8f;
typedef __attribute__((ext_vector_type(4)))  unsigned int u32x4;
typedef __attribute__((ext_vector_type(8)))  int          i32x8;
typedef __attribute__((ext_vector_type(4)))  int          i32x4;

struct __align__(16) U4 { unsigned int x, y, z, w; };  // POD 128-bit chunk (union-safe)

union ABfrag {
    v16bf        v;
    unsigned int u[8];
    U4           q[2];
};

#if defined(__has_builtin)
#if __has_builtin(__builtin_amdgcn_tensor_load_to_lds) && __has_builtin(__builtin_amdgcn_s_wait_tensorcnt)
#define HAVE_TDM 1
#else
#define HAVE_TDM 0
#endif
#else
#define HAVE_TDM 0
#endif

__device__ __forceinline__ float bf2f(unsigned int hbits) {
    return __uint_as_float(hbits << 16);
}
__device__ __forceinline__ unsigned short f2bf(float f) {
    unsigned int u = __float_as_uint(f);
    u += 0x7fffu + ((u >> 16) & 1u);          // round-to-nearest-even
    return (unsigned short)(u >> 16);
}
__device__ __forceinline__ float gelu_f(float x) {
    return 0.5f * x * (1.f + tanhf(0.7978845608028654f * (x + 0.044715f * x * x * x)));
}
__device__ __forceinline__ float sigm_f(float x) {
    return 1.f / (1.f + __expf(-x));
}

#if HAVE_TDM
// ---------------------------------------------------------------------------
// Issue one 2D bf16 tile load via the Tensor Data Mover (wave-level DMA).
// D# per CDNA5 ISA ch.8: group0 = {count|flags, lds_addr, gaddr_lo, gaddr_hi|type},
// group1 packs data_size/pad config, tensor_dim0/1, tile_dim0/1, dim0 stride.
// pad_interval/pad_amount codes reproduce the padded LDS row strides used by
// the WMMA fragment readers (hardware inserts the padding during the DMA).
// ---------------------------------------------------------------------------
__device__ __forceinline__ void tdm_load_2d(
    const void* gaddr, unsigned lds_byte_addr,
    unsigned tile_w_elems, unsigned tile_h_rows,
    unsigned row_stride_elems, unsigned pad_interval_code, unsigned pad_amount_code)
{
    unsigned long long ga = (unsigned long long)gaddr;
    u32x4 g0;
    g0.x = 1u;                                               // count=1, user mode
    g0.y = lds_byte_addr;                                    // LDS dest (bytes)
    g0.z = (unsigned)(ga & 0xffffffffu);                     // global addr lo
    g0.w = (unsigned)((ga >> 32) & 0x01ffffffu) | (2u << 30);// addr hi | type=2
    const unsigned td0 = row_stride_elems;                   // tensor_dim0
    const unsigned td1 = 0x100000u;                          // tensor_dim1 (no OOB)
    i32x8 g1;
    g1[0] = (int)((1u << 16) |                               // data_size = 2 bytes
                  (1u << 20) |                               // pad_enable
                  (pad_interval_code << 22) | (pad_amount_code << 25));
    g1[1] = (int)((td0 & 0xffffu) << 16);                    // tensor_dim0[15:0]
    g1[2] = (int)(((td0 >> 16) & 0xffffu) | ((td1 & 0xffffu) << 16));
    g1[3] = (int)(((td1 >> 16) & 0xffffu) | ((tile_w_elems & 0xffffu) << 16));
    g1[4] = (int)(tile_h_rows & 0xffffu);                    // tile_dim1 (dim2=0)
    g1[5] = (int)row_stride_elems;                           // dim0 stride lo
    g1[6] = 0;
    g1[7] = 0;
    const i32x4 gz = {0, 0, 0, 0};
#if __clang_major__ >= 23
    const i32x8 gz8 = {0, 0, 0, 0, 0, 0, 0, 0};
    __builtin_amdgcn_tensor_load_to_lds(g0, g1, gz, gz, gz8, 0);
#else
    __builtin_amdgcn_tensor_load_to_lds(g0, g1, gz, gz, 0);
#endif
}
__device__ __forceinline__ unsigned lds_offset_of(const void* p) {
    return (unsigned)(unsigned long long)p;   // LDS aperture: low 32 bits = offset
}
#endif

// ---------------------------------------------------------------------------
// generic fp32 -> bf16 convert
// ---------------------------------------------------------------------------
__global__ void cvt_kernel(const float* __restrict__ src, unsigned short* __restrict__ dst, int n) {
    int i = blockIdx.x * blockDim.x + threadIdx.x;
    if (i < n) dst[i] = f2bf(src[i]);
}

// mem_{short,long} [B,Mn,Dn] -> kv_b [B,Sn,Dn] rows [rowOff, rowOff+Mn)
__global__ void cvt2d_kernel(const float* __restrict__ src, unsigned short* __restrict__ dst,
                             int rows, int rowOff) {
    int i = blockIdx.x * blockDim.x + threadIdx.x;
    int total = Bn * rows * Dn;
    if (i >= total) return;
    int c = i & (Dn - 1);
    int r = (i >> 9) % rows;
    int b = i / (rows * Dn);
    dst[((size_t)(b * Sn + rowOff + r)) * Dn + c] = f2bf(src[i]);
}

// ---------------------------------------------------------------------------
// RMS norm: f32 [rows, 512] -> bf16, one row per 512-thread block
// ---------------------------------------------------------------------------
__global__ __launch_bounds__(512) void rms_kernel(const float* __restrict__ xin,
                                                  const float* __restrict__ scale,
                                                  unsigned short* __restrict__ outb) {
    const int row = blockIdx.x;
    const int j = threadIdx.x;
    __shared__ float ws[16];
    __shared__ float rinv;
    float v = xin[(size_t)row * Dn + j];
    float sq = v * v;
    #pragma unroll
    for (int m = 1; m < 32; m <<= 1) sq += __shfl_xor(sq, m, 32);
    if ((j & 31) == 0) ws[j >> 5] = sq;
    __syncthreads();
    if (j == 0) {
        float s = 0.f;
        #pragma unroll
        for (int i = 0; i < 16; ++i) s += ws[i];
        rinv = rsqrtf(s * (1.0f / Dn) + 1e-6f);
    }
    __syncthreads();
    outb[(size_t)row * Dn + j] = f2bf(v * rinv * scale[j]);
}

// ---------------------------------------------------------------------------
// WMMA bf16 GEMM: C[M,N] = act(A[M,K] @ Bw[K,N] + bias) (+ addp)
// block = 128 threads (4 waves), block tile 64x128, wave tile 32x64, K step 32.
// TDM path: per-wave tensor_load_to_lds DMA of next tile slices (double
// buffered, TENSORcnt-synchronized) overlapped with WMMA on current tile.
// ---------------------------------------------------------------------------
#define A_STRIDE 40    // 32 + 8 pad elements  (80B rows,  16B aligned)
#define B_STRIDE 136   // 128 + 8 pad elements (272B rows, 16B aligned)

template <int ACT, bool WF, bool WB, bool WADD>
__global__ __launch_bounds__(128) void gemm_bf16_wmma(
    const unsigned short* __restrict__ A, int lda,
    const unsigned short* __restrict__ Bw, int ldb,
    const float* __restrict__ bias, const float* __restrict__ addp,
    float* __restrict__ Cf, unsigned short* __restrict__ Cb,
    int ldc, int Kdim)
{
    __shared__ __align__(16) unsigned short As[2][64 * A_STRIDE];
    __shared__ __align__(16) unsigned short Bs[2][32 * B_STRIDE];

    const int tid  = threadIdx.x;
    const int lane = tid & 31;
    const int lrow = lane & 15;
    const int lhi  = lane >> 4;
    const int wave = tid >> 5;
    const int wm   = (wave >> 1) * 32;
    const int wn   = (wave & 1) * 64;
    const int m0   = blockIdx.y * 64;
    const int n0   = blockIdx.x * 128;

    const v8f vzero = {0.f, 0.f, 0.f, 0.f, 0.f, 0.f, 0.f, 0.f};
    v8f acc[2][4];
    #pragma unroll
    for (int i = 0; i < 2; ++i)
        #pragma unroll
        for (int j = 0; j < 4; ++j) acc[i][j] = vzero;

    const unsigned short* Abase = A + (size_t)m0 * lda;
    const unsigned short* Bbase = Bw + n0;

    auto compute = [&](const unsigned short* as, const unsigned short* bs) {
        ABfrag af[2], bfr[4];
        #pragma unroll
        for (int i = 0; i < 2; ++i) {
            const int row = wm + i * 16 + lrow;
            #pragma unroll
            for (int jj = 0; jj < 8; ++jj) {
                const int kk = (jj < 4 ? 2 * jj : 2 * jj + 8) + 8 * lhi;  // A-frag K pairs
                af[i].u[jj] = *(const unsigned int*)&as[row * A_STRIDE + kk];
            }
        }
        // B-frag: lane holds N column lrow, K pairs in table order
        #pragma unroll
        for (int i = 0; i < 4; ++i) {
            #pragma unroll
            for (int jj = 0; jj < 8; ++jj) {
                const int kk = (jj < 4 ? 2 * jj : 2 * jj + 8) + 8 * lhi;
                const unsigned short* c = &bs[kk * B_STRIDE + wn + i * 16 + lrow];
                bfr[i].u[jj] = (unsigned int)c[0] | ((unsigned int)c[B_STRIDE] << 16);
            }
        }
        #pragma unroll
        for (int i = 0; i < 2; ++i)
            #pragma unroll
            for (int jj = 0; jj < 4; ++jj)
                acc[i][jj] = __builtin_amdgcn_wmma_f32_16x16x32_bf16(
                    false, af[i].v, false, bfr[jj].v, (short)0, acc[i][jj], false, false);
    };

#if HAVE_TDM
    // ---- TDM double-buffered pipeline: each wave DMAs its slice of A/B ----
    auto issue = [&](int k0, int buf) {
        tdm_load_2d(Abase + (size_t)(wave * 16) * lda + k0,
                    lds_offset_of(&As[buf][wave * 16 * A_STRIDE]),
                    /*w*/32, /*h*/16, /*stride*/(unsigned)lda,
                    /*interval 16DW*/3, /*pad 4DW*/3);
        tdm_load_2d(Bbase + (size_t)(k0 + wave * 8) * ldb,
                    lds_offset_of(&Bs[buf][wave * 8 * B_STRIDE]),
                    /*w*/128, /*h*/8, /*stride*/(unsigned)ldb,
                    /*interval 64DW*/5, /*pad 4DW*/3);
    };
    const int nIt = Kdim >> 5;
    issue(0, 0);
    for (int it = 0; it < nIt; ++it) {
        const int cur = it & 1;
        __builtin_amdgcn_s_wait_tensorcnt(0);   // my slice of buf[cur] landed
        __syncthreads();                        // all slices landed; prev reads done
        if (it + 1 < nIt) issue((it + 1) * 32, cur ^ 1);  // overlap next DMA
        compute(&As[cur][0], &Bs[cur][0]);
    }
#else
    // ---- fallback: synchronous load + ds_store staging ----
    const int arow = tid >> 1, acol = (tid & 1) * 16;
    const int brow = tid >> 2, bcol = (tid & 3) * 32;
    const unsigned short* Aptr = Abase + (size_t)arow * lda + acol;
    const unsigned short* Bptr = Bbase + (size_t)brow * ldb + bcol;
    for (int k0 = 0; k0 < Kdim; k0 += 32) {
        U4 a0 = *(const U4*)(Aptr + k0);
        U4 a1 = *(const U4*)(Aptr + k0 + 8);
        U4 b0 = *(const U4*)(Bptr + (size_t)k0 * ldb);
        U4 b1 = *(const U4*)(Bptr + (size_t)k0 * ldb + 8);
        U4 b2 = *(const U4*)(Bptr + (size_t)k0 * ldb + 16);
        U4 b3 = *(const U4*)(Bptr + (size_t)k0 * ldb + 24);
        if (k0 + 32 < Kdim) {
            __builtin_prefetch(Aptr + k0 + 32, 0, 1);
            __builtin_prefetch(Bptr + (size_t)(k0 + 32) * ldb, 0, 1);
        }
        __syncthreads();
        *(U4*)&As[0][arow * A_STRIDE + acol]     = a0;
        *(U4*)&As[0][arow * A_STRIDE + acol + 8] = a1;
        *(U4*)&Bs[0][brow * B_STRIDE + bcol]      = b0;
        *(U4*)&Bs[0][brow * B_STRIDE + bcol + 8]  = b1;
        *(U4*)&Bs[0][brow * B_STRIDE + bcol + 16] = b2;
        *(U4*)&Bs[0][brow * B_STRIDE + bcol + 24] = b3;
        __syncthreads();
        compute(&As[0][0], &Bs[0][0]);
    }
#endif

    // ---- straight-line epilogue (template-specialized, no runtime branches) ----
    #pragma unroll
    for (int i = 0; i < 2; ++i)
        #pragma unroll
        for (int jj = 0; jj < 4; ++jj) {
            const int col = n0 + wn + jj * 16 + lrow;
            const float bvv = bias[col];
            #pragma unroll
            for (int r = 0; r < 8; ++r) {
                const int row = m0 + wm + i * 16 + r + 8 * lhi;
                float v = acc[i][jj][r] + bvv;
                if constexpr (ACT == 1) v = gelu_f(v);
                const size_t idx = (size_t)row * ldc + col;
                if constexpr (WADD) v += addp[idx];
                if constexpr (WF) Cf[idx] = v;
                if constexpr (WB) Cb[idx] = f2bf(v);
            }
        }
}

// ---------------------------------------------------------------------------
// gamma = clip(sigmoid(gb) * sigmoid(gm), 1e-6, 1-1e-6), in-place into gb
// ---------------------------------------------------------------------------
__global__ void gamma_kernel(float* __restrict__ gb, const float* __restrict__ gm, int n) {
    int i = blockIdx.x * blockDim.x + threadIdx.x;
    if (i >= n) return;
    float g = sigm_f(gb[i]) * sigm_f(gm[i]);
    gb[i] = fminf(1.f - 1e-6f, fmaxf(1e-6f, g));
}

// ---------------------------------------------------------------------------
// sequential recurrence: one block per batch, W_gh register-resident (bf16)
// thread t: output col j = t&511, k-half = t>>9 (256 K each -> 128 packed regs)
// ---------------------------------------------------------------------------
__global__ __launch_bounds__(1024) void rec_kernel(
    const unsigned short* __restrict__ Wgh, const float* __restrict__ b_gh,
    const float* __restrict__ gz, const float* __restrict__ fusedf,
    const float* __restrict__ gamma, const float* __restrict__ normr,
    float* __restrict__ hseq, unsigned short* __restrict__ kvb)
{
    const int b    = blockIdx.x;
    const int t0   = threadIdx.x;
    const int j    = t0 & (Dn - 1);
    const int half = t0 >> 9;

    __shared__ float h_lds[Dn];
    __shared__ float part[2][Dn];
    __shared__ float wsum[16];
    __shared__ float rtot;

    unsigned int wreg[128];                    // 256 bf16 of column j, packed
    #pragma unroll
    for (int p = 0; p < 128; ++p) {
        const int k0 = half * 256 + 2 * p;
        unsigned int lo = Wgh[(size_t)k0 * Dn + j];
        unsigned int hi = Wgh[(size_t)(k0 + 1) * Dn + j];
        wreg[p] = lo | (hi << 16);
    }
    if (t0 < Dn) h_lds[t0] = 0.f;
    __syncthreads();

    for (int t = 0; t < Tn; ++t) {
        float acc = 0.f;
        #pragma unroll
        for (int p = 0; p < 128; ++p) {
            const int k0 = half * 256 + 2 * p;
            acc += h_lds[k0]     * bf2f(wreg[p] & 0xffffu);
            acc += h_lds[k0 + 1] * bf2f(wreg[p] >> 16);
        }
        part[half][j] = acc;
        __syncthreads();

        float hnew = 0.f;
        if (half == 0) {
            const int row = b * Tn + t;
            float s   = part[0][j] + part[1][j] + b_gh[j];
            float gzv = gz[(size_t)row * Dn + j];
            float zt  = fusedf[(size_t)row * (4 * Dn) + 3 * Dn + j];
            float gam = gamma[(size_t)row * Dn + j];
            float hp  = h_lds[j];
            float gate = sigm_f(gzv + s);
            float ht = (1.f - gate) * hp + gate * zt + gam * hp;
            float sq = ht * ht;
            #pragma unroll
            for (int m = 1; m < 32; m <<= 1) sq += __shfl_xor(sq, m, 32);
            if ((t0 & 31) == 0) wsum[t0 >> 5] = sq;
            hnew = ht;
        }
        __syncthreads();
        if (t0 == 0) {
            float s2 = 0.f;
            #pragma unroll
            for (int i = 0; i < 16; ++i) s2 += wsum[i];
            rtot = rsqrtf(s2 * (1.0f / Dn) + 1e-6f);
        }
        __syncthreads();
        if (half == 0) {
            float hn = hnew * rtot * normr[j];
            hn = fminf(100.f, fmaxf(-100.f, hn));
            h_lds[j] = hn;
            const int row = b * Tn + t;
            hseq[(size_t)row * Dn + j] = hn;
            kvb[((size_t)(b * Sn + 2 * Mn + t)) * Dn + j] = f2bf(hn);
        }
        __syncthreads();
    }
}

// ---------------------------------------------------------------------------
// K transpose: kp [B*S, D] bf16 -> kT [B, D, S] bf16 (coalesced writes)
// ---------------------------------------------------------------------------
__global__ void ktrans_kernel(const unsigned short* __restrict__ kp, unsigned short* __restrict__ kT) {
    size_t i = (size_t)blockIdx.x * blockDim.x + threadIdx.x;
    const size_t total = (size_t)Bn * Dn * Sn;
    if (i >= total) return;
    int s = (int)(i % Sn);
    size_t r = i / Sn;
    int d = (int)(r % Dn);
    int b = (int)(r / Dn);
    kT[i] = kp[((size_t)(b * Sn + s)) * Dn + d];
}

// ---------------------------------------------------------------------------
// flash attention with WMMA: one wave (32 threads) per (b, h, 16-row q tile)
// scores via q-frags x kT-frags; online softmax in C-frag layout;
// P through LDS to A-frag layout; P @ V accumulated in 4 C-frags.
// ---------------------------------------------------------------------------
__global__ __launch_bounds__(32) void fattn_kernel(
    const unsigned short* __restrict__ qp,   // [B*T, D] bf16 (q @ Wq + bq)
    const unsigned short* __restrict__ kT,   // [B, D, S] bf16
    const unsigned short* __restrict__ vp,   // [B*S, D] bf16
    unsigned short* __restrict__ att)        // [B*T, D] bf16
{
    const int bid = blockIdx.x;
    const int t16 = bid & 127;
    const int h   = (bid >> 7) & 7;
    const int b   = bid >> 10;
    const int lane = threadIdx.x & 31;
    const int lrow = lane & 15;
    const int lhi  = lane >> 4;

    __shared__ __align__(16) unsigned short Pt[16 * 32];

    // q A-frags (16 x 64 split into two 16 x 32), pre-loaded
    ABfrag qf[2];
    const unsigned short* qbase =
        qp + ((size_t)(b * Tn + t16 * 16 + lrow)) * Dn + h * HDn;
    #pragma unroll
    for (int f = 0; f < 2; ++f)
        #pragma unroll
        for (int jj = 0; jj < 8; ++jj) {
            const int kk = (jj < 4 ? 2 * jj : 2 * jj + 8) + 8 * lhi + f * 32;
            qf[f].u[jj] = *(const unsigned int*)(qbase + kk);
        }

    const v8f vzero = {0.f, 0.f, 0.f, 0.f, 0.f, 0.f, 0.f, 0.f};
    v8f acc[4];
    acc[0] = vzero; acc[1] = vzero; acc[2] = vzero; acc[3] = vzero;
    float mrun[8], lrun[8];
    #pragma unroll
    for (int r = 0; r < 8; ++r) { mrun[r] = -1e30f; lrun[r] = 0.f; }

    const unsigned short* kb = kT + ((size_t)b * Dn + h * HDn) * Sn;

    for (int s0 = 0; s0 < Sn; s0 += 32) {
        if (s0 + 32 < Sn) {   // CDNA5 global_prefetch_b8 for next block
            __builtin_prefetch(kb + (size_t)(lrow + 16 * lhi) * Sn + s0 + 32, 0, 1);
            __builtin_prefetch(vp + ((size_t)(b * Sn + s0 + 32 + lrow + 16 * lhi)) * Dn + h * HDn, 0, 1);
        }
        // K^T B-frags: [f = hd chunk of 32][nc = s chunk of 16]
        ABfrag bk[2][2];
        #pragma unroll
        for (int f = 0; f < 2; ++f)
            #pragma unroll
            for (int nc = 0; nc < 2; ++nc)
                #pragma unroll
                for (int jj = 0; jj < 8; ++jj) {
                    const int hd = f * 32 + (jj < 4 ? 2 * jj : 2 * jj + 8) + 8 * lhi;
                    const unsigned short* p = kb + (size_t)hd * Sn + s0 + nc * 16 + lrow;
                    bk[f][nc].u[jj] = (unsigned int)p[0] | ((unsigned int)p[Sn] << 16);
                }
        v8f s0f = vzero, s1f = vzero;
        s0f = __builtin_amdgcn_wmma_f32_16x16x32_bf16(false, qf[0].v, false, bk[0][0].v, (short)0, s0f, false, false);
        s0f = __builtin_amdgcn_wmma_f32_16x16x32_bf16(false, qf[1].v, false, bk[1][0].v, (short)0, s0f, false, false);
        s1f = __builtin_amdgcn_wmma_f32_16x16x32_bf16(false, qf[0].v, false, bk[0][1].v, (short)0, s1f, false, false);
        s1f = __builtin_amdgcn_wmma_f32_16x16x32_bf16(false, qf[1].v, false, bk[1][1].v, (short)0, s1f, false, false);

        float p0[8], p1[8];
        #pragma unroll
        for (int r = 0; r < 8; ++r) {
            s0f[r] *= 0.125f;                          // 1/sqrt(HD)
            s1f[r] *= 0.125f;
            float t = fmaxf(s0f[r], s1f[r]);
            #pragma unroll
            for (int m = 1; m < 16; m <<= 1) t = fmaxf(t, __shfl_xor(t, m, 32));
            const float mnew  = fmaxf(mrun[r], t);
            const float alpha = __expf(mrun[r] - mnew);
            p0[r] = __expf(s0f[r] - mnew);
            p1[r] = __expf(s1f[r] - mnew);
            float rs = p0[r] + p1[r];
            #pragma unroll
            for (int m = 1; m < 16; m <<= 1) rs += __shfl_xor(rs, m, 32);
            lrun[r] = lrun[r] * alpha + rs;
            mrun[r] = mnew;
            acc[0][r] *= alpha; acc[1][r] *= alpha; acc[2][r] *= alpha; acc[3][r] *= alpha;
        }

        // P (C-frag layout) -> LDS row-major 16x32 bf16
        #pragma unroll
        for (int r = 0; r < 8; ++r) {
            const int row = r + 8 * lhi;
            Pt[row * 32 + lrow]      = f2bf(p0[r]);
            Pt[row * 32 + lrow + 16] = f2bf(p1[r]);
        }
        __syncthreads();   // single-wave WG: ds-count fence

        // P as A-frag 16x32
        ABfrag pf;
        #pragma unroll
        for (int jj = 0; jj < 8; ++jj) {
            const int kk = (jj < 4 ? 2 * jj : 2 * jj + 8) + 8 * lhi;
            pf.u[jj] = *(const unsigned int*)&Pt[lrow * 32 + kk];
        }
        // V B-frags: lane holds hd column lrow, s pairs in table order
        const unsigned short* vb =
            vp + ((size_t)(b * Sn + s0)) * Dn + h * HDn + lrow;
        #pragma unroll
        for (int nf = 0; nf < 4; ++nf) {
            ABfrag vf;
            #pragma unroll
            for (int jj = 0; jj < 8; ++jj) {
                const int kk = (jj < 4 ? 2 * jj : 2 * jj + 8) + 8 * lhi;
                const unsigned short* p = vb + (size_t)kk * Dn + nf * 16;
                vf.u[jj] = (unsigned int)p[0] | ((unsigned int)p[Dn] << 16);
            }
            acc[nf] = __builtin_amdgcn_wmma_f32_16x16x32_bf16(
                false, pf.v, false, vf.v, (short)0, acc[nf], false, false);
        }
        __syncthreads();   // Pt reused next iteration
    }

    #pragma unroll
    for (int nf = 0; nf < 4; ++nf)
        #pragma unroll
        for (int r = 0; r < 8; ++r) {
            const int row = r + 8 * lhi;
            const float o = acc[nf][r] / fmaxf(lrun[r], 1e-20f);
            att[((size_t)(b * Tn + t16 * 16 + row)) * Dn + h * HDn + nf * 16 + lrow] = f2bf(o);
        }
}

// ---------------------------------------------------------------------------
// x_res = x + h_seq + att_o + 0.1 * v   (v = fused[..., 2D:3D])
// ---------------------------------------------------------------------------
__global__ void resid_kernel(const float* __restrict__ x, const float* __restrict__ hseq,
                             const float* __restrict__ atto, const float* __restrict__ fusedf,
                             float* __restrict__ xres, int n) {
    int i = blockIdx.x * blockDim.x + threadIdx.x;
    if (i >= n) return;
    const int row = i >> 9, col = i & (Dn - 1);
    const float v = fusedf[(size_t)row * (4 * Dn) + 2 * Dn + col];
    xres[i] = x[i] + hseq[i] + atto[i] + 0.1f * v;
}

// ---------------------------------------------------------------------------
extern "C" void kernel_launch(void* const* d_in, const int* in_sizes, int n_in,
                              void* d_out, int out_size, void* d_ws, size_t ws_size,
                              hipStream_t stream) {
    const float* x      = (const float*)d_in[0];
    const float* mem_s  = (const float*)d_in[1];
    const float* mem_l  = (const float*)d_in[2];
    const float* n1s    = (const float*)d_in[3];
    const float* n2s    = (const float*)d_in[4];
    const float* nrs    = (const float*)d_in[5];
    const float* W_fused = (const float*)d_in[6];  const float* b_fused = (const float*)d_in[7];
    const float* W_gz    = (const float*)d_in[8];  const float* b_gz    = (const float*)d_in[9];
    const float* W_gh    = (const float*)d_in[10]; const float* b_gh    = (const float*)d_in[11];
    const float* W_g1    = (const float*)d_in[12]; const float* b_g1    = (const float*)d_in[13];
    const float* W_g2    = (const float*)d_in[14]; const float* b_g2    = (const float*)d_in[15];
    const float* W_c1    = (const float*)d_in[16]; const float* b_c1    = (const float*)d_in[17];
    const float* W_c2    = (const float*)d_in[18]; const float* b_c2    = (const float*)d_in[19];
    const float* Wq      = (const float*)d_in[20]; const float* bq      = (const float*)d_in[21];
    const float* Wk      = (const float*)d_in[22]; const float* bk      = (const float*)d_in[23];
    const float* Wv      = (const float*)d_in[24]; const float* bv      = (const float*)d_in[25];
    const float* Wo      = (const float*)d_in[26]; const float* bo      = (const float*)d_in[27];
    const float* W_ffn1  = (const float*)d_in[28]; const float* b_ffn1  = (const float*)d_in[29];
    const float* W_ffn2  = (const float*)d_in[30]; const float* b_ffn2  = (const float*)d_in[31];

    char* base = (char*)d_ws;
    size_t off = 0;
    auto alloc = [&](size_t elems, size_t esz) -> char* {
        char* p = base + off;
        off = (off + elems * esz + 255) & ~(size_t)255;
        return p;
    };

    // bf16 weights
    unsigned short* wfused_b = (unsigned short*)alloc((size_t)Dn * 4 * Dn, 2);
    unsigned short* wgz_b    = (unsigned short*)alloc((size_t)Dn * Dn, 2);
    unsigned short* wgh_b    = (unsigned short*)alloc((size_t)Dn * Dn, 2);
    unsigned short* wg1_b    = (unsigned short*)alloc((size_t)Dn * DQn, 2);
    unsigned short* wg2_b    = (unsigned short*)alloc((size_t)DQn * Dn, 2);
    unsigned short* wc1_b    = (unsigned short*)alloc((size_t)Dn * DQn, 2);
    unsigned short* wc2_b    = (unsigned short*)alloc((size_t)DQn * Dn, 2);
    unsigned short* wq_b     = (unsigned short*)alloc((size_t)Dn * Dn, 2);
    unsigned short* wk_b     = (unsigned short*)alloc((size_t)Dn * Dn, 2);
    unsigned short* wv_b     = (unsigned short*)alloc((size_t)Dn * Dn, 2);
    unsigned short* wo_b     = (unsigned short*)alloc((size_t)Dn * Dn, 2);
    unsigned short* wffn1_b  = (unsigned short*)alloc((size_t)Dn * Fn, 2);
    unsigned short* wffn2_b  = (unsigned short*)alloc((size_t)Fn * Dn, 2);
    // bf16 activations
    unsigned short* xn_b    = (unsigned short*)alloc((size_t)BT * Dn, 2);
    unsigned short* fused_b = (unsigned short*)alloc((size_t)BT * 4 * Dn, 2);
    unsigned short* g1_b    = (unsigned short*)alloc((size_t)BT * DQn, 2);
    unsigned short* c1_b    = (unsigned short*)alloc((size_t)BT * DQn, 2);
    unsigned short* kv_b    = (unsigned short*)alloc((size_t)BS * Dn, 2);
    unsigned short* qp_b    = (unsigned short*)alloc((size_t)BT * Dn, 2);
    unsigned short* kp_b    = (unsigned short*)alloc((size_t)BS * Dn, 2);
    unsigned short* kT_b    = (unsigned short*)alloc((size_t)BS * Dn, 2);
    unsigned short* vp_b    = (unsigned short*)alloc((size_t)BS * Dn, 2);
    unsigned short* att_b   = (unsigned short*)alloc((size_t)BT * Dn, 2);
    unsigned short* xn2_b   = (unsigned short*)alloc((size_t)BT * Dn, 2);
    unsigned short* ffn1_b  = (unsigned short*)alloc((size_t)BT * Fn, 2);
    // f32 activations
    float* fused_f = (float*)alloc((size_t)BT * 4 * Dn, 4);
    float* gb_f    = (float*)alloc((size_t)BT * Dn, 4);   // -> gamma (in place)
    float* gm_f    = (float*)alloc((size_t)BT * Dn, 4);
    float* gz_f    = (float*)alloc((size_t)BT * Dn, 4);
    float* hseq_f  = (float*)alloc((size_t)BT * Dn, 4);
    float* atto_f  = (float*)alloc((size_t)BT * Dn, 4);
    float* xres_f  = (float*)alloc((size_t)BT * Dn, 4);
    (void)ws_size; (void)in_sizes; (void)n_in; (void)out_size;

    auto cvt = [&](const float* s, unsigned short* d, int n) {
        cvt_kernel<<<(n + 255) / 256, 256, 0, stream>>>(s, d, n);
    };
    // dispatch to template-specialized GEMM (act: 0=none, 1=gelu)
    auto gemm = [&](int M, int N, int K, const unsigned short* A, int lda,
                    const unsigned short* Bw, int ldb, const float* bias,
                    const float* addp, float* Cf, unsigned short* Cb, int ldc, int act) {
        dim3 g(N / 128, M / 64);
        if (act == 0 && Cf && Cb)
            gemm_bf16_wmma<0, true, true, false><<<g, 128, 0, stream>>>(A, lda, Bw, ldb, bias, addp, Cf, Cb, ldc, K);
        else if (act == 0 && Cf && addp)
            gemm_bf16_wmma<0, true, false, true><<<g, 128, 0, stream>>>(A, lda, Bw, ldb, bias, addp, Cf, Cb, ldc, K);
        else if (act == 0 && Cf)
            gemm_bf16_wmma<0, true, false, false><<<g, 128, 0, stream>>>(A, lda, Bw, ldb, bias, addp, Cf, Cb, ldc, K);
        else if (act == 0)
            gemm_bf16_wmma<0, false, true, false><<<g, 128, 0, stream>>>(A, lda, Bw, ldb, bias, addp, Cf, Cb, ldc, K);
        else
            gemm_bf16_wmma<1, false, true, false><<<g, 128, 0, stream>>>(A, lda, Bw, ldb, bias, addp, Cf, Cb, ldc, K);
    };

    // ---- weight conversion ----
    cvt(W_fused, wfused_b, Dn * 4 * Dn);
    cvt(W_gz, wgz_b, Dn * Dn);
    cvt(W_gh, wgh_b, Dn * Dn);
    cvt(W_g1, wg1_b, Dn * DQn);
    cvt(W_g2, wg2_b, DQn * Dn);
    cvt(W_c1, wc1_b, Dn * DQn);
    cvt(W_c2, wc2_b, DQn * Dn);
    cvt(Wq, wq_b, Dn * Dn);
    cvt(Wk, wk_b, Dn * Dn);
    cvt(Wv, wv_b, Dn * Dn);
    cvt(Wo, wo_b, Dn * Dn);
    cvt(W_ffn1, wffn1_b, Dn * Fn);
    cvt(W_ffn2, wffn2_b, Fn * Dn);
    // memories into kv buffer
    cvt2d_kernel<<<(Bn * Mn * Dn + 255) / 256, 256, 0, stream>>>(mem_s, kv_b, Mn, 0);
    cvt2d_kernel<<<(Bn * Mn * Dn + 255) / 256, 256, 0, stream>>>(mem_l, kv_b, Mn, Mn);

    // ---- norm1 + fused projection ----
    rms_kernel<<<BT, 512, 0, stream>>>(x, n1s, xn_b);
    gemm(BT, 4 * Dn, Dn, xn_b, Dn, wfused_b, 4 * Dn, b_fused, nullptr, fused_f, fused_b, 4 * Dn, 0);

    // ---- gating MLPs + input gate logits ----
    gemm(BT, DQn, Dn, fused_b + 3 * Dn, 4 * Dn, wg1_b, DQn, b_g1, nullptr, nullptr, g1_b, DQn, 1);
    gemm(BT, Dn, DQn, g1_b, DQn, wg2_b, Dn, b_g2, nullptr, gb_f, nullptr, Dn, 0);
    gemm(BT, DQn, Dn, fused_b + Dn, 4 * Dn, wc1_b, DQn, b_c1, nullptr, nullptr, c1_b, DQn, 1);
    gemm(BT, Dn, DQn, c1_b, DQn, wc2_b, Dn, b_c2, nullptr, gm_f, nullptr, Dn, 0);
    gemm(BT, Dn, Dn, fused_b + 3 * Dn, 4 * Dn, wgz_b, Dn, b_gz, nullptr, gz_f, nullptr, Dn, 0);
    gamma_kernel<<<(BT * Dn + 255) / 256, 256, 0, stream>>>(gb_f, gm_f, BT * Dn);

    // ---- sequential recurrence (W_gh register-resident) ----
    rec_kernel<<<Bn, 1024, 0, stream>>>(wgh_b, b_gh, gz_f, fused_f, gb_f, nrs, hseq_f, kv_b);

    // ---- attention projections ----
    gemm(BT, Dn, Dn, fused_b, 4 * Dn, wq_b, Dn, bq, nullptr, nullptr, qp_b, Dn, 0);
    gemm(BS, Dn, Dn, kv_b, Dn, wk_b, Dn, bk, nullptr, nullptr, kp_b, Dn, 0);
    gemm(BS, Dn, Dn, kv_b, Dn, wv_b, Dn, bv, nullptr, nullptr, vp_b, Dn, 0);
    ktrans_kernel<<<((size_t)Bn * Dn * Sn + 255) / 256, 256, 0, stream>>>(kp_b, kT_b);

    // ---- flash attention (WMMA) ----
    fattn_kernel<<<Bn * Hn * (Tn / 16), 32, 0, stream>>>(qp_b, kT_b, vp_b, att_b);
    gemm(BT, Dn, Dn, att_b, Dn, wo_b, Dn, bo, nullptr, atto_f, nullptr, Dn, 0);

    // ---- residual, norm2, FFN (final GEMM fuses + x_res, writes d_out) ----
    resid_kernel<<<(BT * Dn + 255) / 256, 256, 0, stream>>>(x, hseq_f, atto_f, fused_f, xres_f, BT * Dn);
    rms_kernel<<<BT, 512, 0, stream>>>(xres_f, n2s, xn2_b);
    gemm(BT, Fn, Dn, xn2_b, Dn, wffn1_b, Fn, b_ffn1, nullptr, nullptr, ffn1_b, Fn, 1);
    gemm(BT, Dn, Fn, ffn1_b, Fn, wffn2_b, Dn, b_ffn2, xres_f, (float*)d_out, nullptr, Dn, 0);
}
